// L1AttnSparse_54090818126481
// MI455X (gfx1250) — matrix operand order repaired and checked by reading, not compiled, not run on hardware
//
#include <hip/hip_runtime.h>
#include <math.h>

// L1 sparse attention for MI455X (gfx1250, wave32).
// Scores s = -scale*L1dist are always <= 0, so softmax with a fixed shift of 0
// is exact (shift-invariant, exp(s) in (0,1], no overflow possible).
// P@V runs on v_wmma_f32_16x16x32_f16 with f32 accumulation.
// Host pre-folds -scale*log2(e) so each score costs 1 mul + 1 v_exp_f32.

typedef __attribute__((ext_vector_type(16))) _Float16 v16h;
typedef __attribute__((ext_vector_type(8)))  float    v8f;

#define BS_    2
#define NCTX_  2048
#define H_     8
#define W_     32
#define NT_    512
#define CHUNK_ 32
#define NCHUNKS_ (NT_ / CHUNK_)      // 16
#define ROWS_PER_WAVE_ 16
#define WAVES_ 4
#define ROWS_PER_BLOCK_ (ROWS_PER_WAVE_ * WAVES_)  // 64
// LDS: qg f32 [NT][W] = 64KB ; V pre-swizzled f16 B tiles = 32KB
#define LDS_Q_BYTES (NT_ * W_ * 4)
#define LDS_VB_BYTES (NCHUNKS_ * 2 * 32 * 16 * 2)
#define LDS_TOTAL (LDS_Q_BYTES + LDS_VB_BYTES)      // 98304

__global__ void L1AttnSparse_54090818126481_kernel(
    const float* __restrict__ q, const float* __restrict__ k,
    const float* __restrict__ v, const int* __restrict__ indx,
    float* __restrict__ out, float scale2 /* = -scale*log2(e) */)
{
  extern __shared__ char smem[];
  float*    lds_q  = (float*)smem;                      // [NT][W] f32
  _Float16* lds_vB = (_Float16*)(smem + LDS_Q_BYTES);   // [chunk][ntile][lane][16] f16

  const int tid  = threadIdx.x;       // 0..127
  const int lane = tid & 31;
  const int wave = tid >> 5;          // 0..3
  const int bb   = blockIdx.z;
  const int hh   = blockIdx.y;
  const int tg   = blockIdx.x;        // row group 0..7
  const int half = lane >> 4;         // 0/1
  const int m    = lane & 15;

  // ---- setup: gather q_g rows into LDS (f32, float4 segments) ----
  for (int s = tid; s < NT_ * 8; s += blockDim.x) {
    int j = s >> 3, p = s & 7;
    int row = indx[j];
    const float4* src =
        (const float4*)(q + (((long long)bb * NCTX_ + row) * H_ + hh) * W_) + p;
    ((float4*)(lds_q + j * W_))[p] = *src;
  }

  // ---- setup: gather v_g straight into WMMA-B-swizzled f16 layout ----
  // B 32x16 f16 layout (wave32): lane holds column n=lane&15;
// lanes 0-15 hold K={0..7,16..23}, lanes 16-31 hold K={8..15,24..31} (2 f16/VGPR).
  for (int s = tid; s < NCHUNKS_ * 2 * 32; s += blockDim.x) {
    int c     = s >> 6;
    int r     = s & 63;
    int ntile = r >> 5;
    int bl    = r & 31;
    int kbase = bl & 16;
    int ncol  = ntile * 16 + (bl & 15);
    union { _Float16 h[16]; uint4 u[2]; } buf;
#pragma unroll
    for (int t = 0; t < 16; ++t) {
      int row = indx[c * CHUNK_ + (kbase >> 1) + ((t < 8) ? t : t + 8)];
      buf.h[t] = (_Float16)v[(((long long)bb * NCTX_ + row) * H_ + hh) * W_ + ncol];
    }
    uint4* dst = (uint4*)(lds_vB + (size_t)s * 16);
    dst[0] = buf.u[0];
    dst[1] = buf.u[1];
  }

  __syncthreads();

  // ---- register-cache this lane's K row (row i0+m, both halves duplicate) ----
  const int i0 = tg * ROWS_PER_BLOCK_ + wave * ROWS_PER_WAVE_;
  int krow = indx[i0 + m];
  float4 kreg[8];
  const float4* ksrc =
      (const float4*)(k + (((long long)bb * NCTX_ + krow) * H_ + hh) * W_);
#pragma unroll
  for (int p = 0; p < 8; ++p) kreg[p] = ksrc[p];

  // ---- softmax numerator + WMMA over 16 column chunks (shift = 0, exact) ----
  v8f acc0 = {}; v8f acc1 = {};
  float lsum = 0.0f;                 // this lane's partial sum of exp(s)
  const int kA = half * 8;           // A-matrix K base: {kA+0..7, kA+16..23}

  for (int c = 0; c < NCHUNKS_; ++c) {
    // issue B-tile LDS loads early; the ~16*8 VALU score phase below hides
    // the DS latency before the WMMAs consume them.
    v16h bv0 = *(const v16h*)(lds_vB + ((size_t)(c * 2 + 0) * 32 + lane) * 16);
    v16h bv1 = *(const v16h*)(lds_vB + ((size_t)(c * 2 + 1) * 32 + lane) * 16);

    v16h a;
#pragma unroll
    for (int t = 0; t < 16; ++t) {
      int j = c * CHUNK_ + kA + ((t < 8) ? t : t + 8);
      const float4* qrow = (const float4*)(lds_q + j * W_);  // broadcast within half
      float s = 0.0f;
#pragma unroll
      for (int p = 0; p < 8; ++p) {
        float4 qv = qrow[p];
        s += fabsf(kreg[p].x - qv.x);
        s += fabsf(kreg[p].y - qv.y);
        s += fabsf(kreg[p].z - qv.z);
        s += fabsf(kreg[p].w - qv.w);
      }
      float pv = __builtin_amdgcn_exp2f(scale2 * s);  // exp(-scale*s) in (0,1]
      lsum += pv;
      a[t] = (_Float16)pv;             // element order == A-matrix VGPR layout
    }

    acc0 = __builtin_amdgcn_wmma_f32_16x16x32_f16(false, a, false, bv0,
                                                  (short)0, acc0, false, false);
    acc1 = __builtin_amdgcn_wmma_f32_16x16x32_f16(false, a, false, bv1,
                                                  (short)0, acc1, false, false);
  }

  // combine the two lane-halves of each row's denominator (one shuffle, once)
  float rowsum = lsum + __shfl_xor(lsum, 16, 32);

  // ---- normalize by row sum and store (C/D layout: VGPR rr -> row rr+8*half,
  //      lane&15 -> column). rowsum for row mm lives in lanes mm and mm+16. ----
#pragma unroll
  for (int rr = 0; rr < 8; ++rr) {
    float rs  = __shfl(rowsum, rr + half * 8, 32);
    float inv = __builtin_amdgcn_rcpf(rs);   // softmax normalize: v_rcp_f32
    int   mi  = rr + half * 8;
    float* op = out + (((long long)bb * NT_ + (i0 + mi)) * H_ + hh) * W_;
    op[m]      = acc0[rr] * inv;
    op[m + 16] = acc1[rr] * inv;
  }
}

extern "C" void kernel_launch(void* const* d_in, const int* in_sizes, int n_in,
                              void* d_out, int out_size, void* d_ws, size_t ws_size,
                              hipStream_t stream) {
  const float* q    = (const float*)d_in[0];
  const float* k    = (const float*)d_in[1];
  const float* v    = (const float*)d_in[2];
  const int*   indx = (const int*)d_in[3];
  float*       out  = (float*)d_out;
  (void)d_ws; (void)ws_size; (void)n_in; (void)out_size;

  int ntok = in_sizes[3];               // 512
  float scale  = 1.0f / sqrtf((float)ntok);
  float scale2 = -scale * 1.4426950408889634f;   // fold log2(e) for v_exp_f32

  dim3 grid(NT_ / ROWS_PER_BLOCK_, H_, BS_);   // (8, 8, 2)
  dim3 block(WAVES_ * 32);                     // 128 threads, 4 waves
  hipLaunchKernelGGL(L1AttnSparse_54090818126481_kernel, grid, block,
                     LDS_TOTAL, stream, q, k, v, indx, out, scale2);
}
